// RoPEAttention_28449863369064
// MI455X (gfx1250) — compile-verified, matrix operand order checked by reading, not run on hardware
//
#include <hip/hip_runtime.h>

// ---------------------------------------------------------------------------
// RoPE attention for MI455X (gfx1250, wave32, WMMA bf16 16x16x32).
// B=2, N=2048, C=1024, H=16, D=64.
// Pipeline: [QKV GEMM + bias + RoPE (+q pre-scale)] -> [flash attention]
//           -> [out-proj GEMM].
// GEMM block tile 128x128, 8 waves (4x2), 32x64 per wave: 8 WMMAs per 6
// fragment loads per K-step (4/3 the WMMA:LDS ratio of a 2x2 wave tile).
// ---------------------------------------------------------------------------

constexpr int kB = 2;
constexpr int kN = 2048;
constexpr int kC = 1024;
constexpr int kH = 16;
constexpr int kD = 64;
constexpr int kM = kB * kN;          // 4096 tokens

typedef __bf16 v8bf  __attribute__((ext_vector_type(8)));
typedef __bf16 v16bf __attribute__((ext_vector_type(16)));
typedef float  v8f   __attribute__((ext_vector_type(8)));

// WMMA wrapper: D = A(16x32 bf16) * B(32x16 bf16) + C(16x16 f32)
__device__ __forceinline__ v8f wmma_bf16(v16bf a, v16bf b, v8f c) {
  return __builtin_amdgcn_wmma_f32_16x16x32_bf16(
      /*neg_a=*/false, a, /*neg_b=*/false, b,
      /*c_mod=*/(short)0, c, /*reuse_a=*/false, /*reuse_b=*/false);
}

// Per-lane A/B fragment load from a row-major [16][>=32] bf16 tile row.
// ISA layout: lane<16 holds K {0..7, 16..23}; lane>=16 holds K {8..15, 24..31}.
// `row` must point at the tile row for (lane & 15), 16-byte aligned.
__device__ __forceinline__ v16bf load_frag(const __bf16* row, int lane) {
  const int base = (lane >> 4) << 3;                 // 0 or 8
  v8bf lo = *(const v8bf*)(row + base);
  v8bf hi = *(const v8bf*)(row + base + 16);
  return __builtin_shufflevector(lo, hi, 0, 1, 2, 3, 4, 5, 6, 7,
                                 8, 9, 10, 11, 12, 13, 14, 15);
}

__device__ __forceinline__ float rowmax16(float v) {
  v = fmaxf(v, __shfl_xor(v, 1, 16));
  v = fmaxf(v, __shfl_xor(v, 2, 16));
  v = fmaxf(v, __shfl_xor(v, 4, 16));
  v = fmaxf(v, __shfl_xor(v, 8, 16));
  return v;
}
__device__ __forceinline__ float rowsum16(float v) {
  v += __shfl_xor(v, 1, 16);
  v += __shfl_xor(v, 2, 16);
  v += __shfl_xor(v, 4, 16);
  v += __shfl_xor(v, 8, 16);
  return v;
}

// Stage one 128x32 fp32 panel -> bf16 LDS tile (16 elems / thread, 256 thr).
__device__ __forceinline__ void stage_f32(__bf16 (*tile)[40],
                                          const float* __restrict__ src,
                                          int ldK, int k0, int tid,
                                          bool prefetch) {
  const int row = tid >> 1;            // 0..127
  const int c16 = (tid & 1) << 4;      // 0 or 16
  const float4* p = (const float4*)(src + (size_t)row * ldK + k0 + c16);
  float4 f0 = p[0], f1 = p[1], f2 = p[2], f3 = p[3];
  v8bf lo = {(__bf16)f0.x, (__bf16)f0.y, (__bf16)f0.z, (__bf16)f0.w,
             (__bf16)f1.x, (__bf16)f1.y, (__bf16)f1.z, (__bf16)f1.w};
  v8bf hi = {(__bf16)f2.x, (__bf16)f2.y, (__bf16)f2.z, (__bf16)f2.w,
             (__bf16)f3.x, (__bf16)f3.y, (__bf16)f3.z, (__bf16)f3.w};
  *(v8bf*)&tile[row][c16]     = lo;
  *(v8bf*)&tile[row][c16 + 8] = hi;
  if (prefetch)
    __builtin_prefetch(src + (size_t)row * ldK + k0 + 32 + c16, 0, 1);
}

// ---------------------------------------------------------------------------
// Kernel 1: QKV = x @ w_qkv^T + b, fused RoPE on Q/K (q also pre-scaled by
// 1/sqrt(D)), bf16 [B,H,N,D] outputs.
// ---------------------------------------------------------------------------
__global__ __launch_bounds__(256)
void qkv_rope_kernel(const float* __restrict__ x, const float* __restrict__ w,
                     const float* __restrict__ bias,
                     const float* __restrict__ fc, const float* __restrict__ fs,
                     __bf16* __restrict__ qws, __bf16* __restrict__ kws,
                     __bf16* __restrict__ vws) {
  __shared__ __align__(16) __bf16 As[128][40];   // padded: no bank conflicts
  __shared__ __align__(16) __bf16 Bs[128][40];

  const int tid  = threadIdx.x;
  const int lane = tid & 31, wid = tid >> 5;
  const int wm = wid & 3, wn = wid >> 2;         // 4x2 wave grid
  const int rowBase = blockIdx.x * 128;
  const int colBase = blockIdx.y * 128;
  const int K = kC;

  v8f acc[2][4] = {};

  for (int k0 = 0; k0 < K; k0 += 32) {
    stage_f32(As, x + (size_t)rowBase * K, K, k0, tid, k0 + 32 < K);
    stage_f32(Bs, w + (size_t)colBase * K, K, k0, tid, k0 + 32 < K);
    __syncthreads();
    v16bf af[2], bf[4];
#pragma unroll
    for (int i = 0; i < 2; ++i)
      af[i] = load_frag(&As[wm * 32 + i * 16 + (lane & 15)][0], lane);
#pragma unroll
    for (int j = 0; j < 4; ++j)
      bf[j] = load_frag(&Bs[wn * 64 + j * 16 + (lane & 15)][0], lane);
#pragma unroll
    for (int i = 0; i < 2; ++i)
#pragma unroll
      for (int j = 0; j < 4; ++j)
        acc[i][j] = wmma_bf16(af[i], bf[j], acc[i][j]);
    __syncthreads();
  }

  // Epilogue: bias + RoPE (Q,K) / passthrough (V); scatter to [B,H,N,D] bf16.
  const int hhalf = (lane >> 4) << 3;
#pragma unroll
  for (int i = 0; i < 2; ++i) {
#pragma unroll
    for (int j = 0; j < 4; ++j) {
      const int cg  = colBase + wn * 64 + j * 16 + (lane & 15);
      const int s   = cg >> 10;          // 0=q 1=k 2=v (uniform per tile)
      const int rem = cg & 1023;
      const int h   = rem >> 6;          // uniform per tile
      const int d   = rem & 63;          // per-lane
#pragma unroll
      for (int r = 0; r < 8; ++r) {
        const int mg = rowBase + wm * 32 + i * 16 + r + hhalf;
        const int bt = mg >> 11;         // / kN
        const int nt = mg & (kN - 1);
        float val = acc[i][j][r] + bias[cg];
        const size_t oidx = (((size_t)bt * kH + h) * kN + nt) * kD + d;
        if (s == 2) {
          vws[oidx] = (__bf16)val;
        } else {
          // even/odd d live in adjacent lanes -> pair exchange
          float partner = __shfl_xor(val, 1, 32);
          float cs = fc[nt * (kD / 2) + (d >> 1)];
          float sn = fs[nt * (kD / 2) + (d >> 1)];
          float o  = (d & 1) ? (partner * sn + val * cs)
                             : (val * cs - partner * sn);
          if (s == 0) o *= 0.125f;       // fold 1/sqrt(D) into q once
          ((s == 0) ? qws : kws)[oidx] = (__bf16)o;
        }
      }
    }
  }
}

// ---------------------------------------------------------------------------
// Kernel 2: flash attention. Grid (N/128, B*H); 8 waves, 16 q-rows per wave.
// Key blocks of 32 (one WMMA K-step for P@V). V staged transposed in LDS.
// Q already pre-scaled by 1/sqrt(D).
// ---------------------------------------------------------------------------
__global__ __launch_bounds__(256)
void flash_attn_kernel(const __bf16* __restrict__ qws,
                       const __bf16* __restrict__ kws,
                       const __bf16* __restrict__ vws,
                       __bf16* __restrict__ aows) {
  __shared__ __align__(16) __bf16 Ks[32][72];        // [key][d], padded
  __shared__ __align__(16) __bf16 Vt[64][40];        // [d][key], padded
  __shared__ __align__(16) __bf16 Ps[8][16][40];     // per-wave P scratch

  const int tid  = threadIdx.x;
  const int lane = tid & 31, wid = tid >> 5;
  const int bh   = blockIdx.y;
  const int bb   = bh / kH, hh = bh % kH;
  const int qBase = blockIdx.x * 128;

  const __bf16* qbase = qws + (((size_t)bb * kH + hh) * kN) * kD;
  const __bf16* kbase = kws + (((size_t)bb * kH + hh) * kN) * kD;
  const __bf16* vbase = vws + (((size_t)bb * kH + hh) * kN) * kD;

  // Q fragments for this wave's 16 rows (D split 0..31 / 32..63), loaded once.
  const __bf16* qrow = qbase + (size_t)(qBase + wid * 16 + (lane & 15)) * kD;
  const v16bf q0 = load_frag(qrow, lane);
  const v16bf q1 = load_frag(qrow + 32, lane);

  v8f o[4] = {};
  float mrow[8], lrow[8];
#pragma unroll
  for (int r = 0; r < 8; ++r) { mrow[r] = -3.0e38f; lrow[r] = 0.0f; }

  for (int kb = 0; kb < kN / 32; ++kb) {
    __syncthreads();
    {  // stage K (row-major) and V (transposed), 8 bf16 per thread each
      const int row = tid >> 3;           // key 0..31
      const int c8  = (tid & 7) << 3;     // d chunk
      const v8bf kv = *(const v8bf*)(kbase + (size_t)(kb * 32 + row) * kD + c8);
      *(v8bf*)&Ks[row][c8] = kv;
      const v8bf vv = *(const v8bf*)(vbase + (size_t)(kb * 32 + row) * kD + c8);
#pragma unroll
      for (int i = 0; i < 8; ++i) Vt[c8 + i][row] = vv[i];
    }
    __syncthreads();

    // scores: S[16q x 32k] = Q(16x64) @ K^T(64x32) -> two 16x16 accumulators
    v8f s0 = {}, s1 = {};
    {
      v16bf k0a = load_frag(&Ks[(lane & 15)][0], lane);
      v16bf k0b = load_frag(&Ks[(lane & 15)][32], lane);
      v16bf k1a = load_frag(&Ks[16 + (lane & 15)][0], lane);
      v16bf k1b = load_frag(&Ks[16 + (lane & 15)][32], lane);
      s0 = wmma_bf16(q0, k0a, s0);
      s0 = wmma_bf16(q1, k0b, s0);
      s1 = wmma_bf16(q0, k1a, s1);
      s1 = wmma_bf16(q1, k1b, s1);
    }

    // online softmax per row, rescale running output, emit P to LDS (bf16)
#pragma unroll
    for (int r = 0; r < 8; ++r) {
      float sm0 = s0[r], sm1 = s1[r];    // q was pre-scaled
      float mnew = fmaxf(mrow[r], rowmax16(fmaxf(sm0, sm1)));
      float alpha = __expf(mrow[r] - mnew);
      float p0 = __expf(sm0 - mnew);
      float p1 = __expf(sm1 - mnew);
      lrow[r] = lrow[r] * alpha + rowsum16(p0 + p1);
      mrow[r] = mnew;
#pragma unroll
      for (int t = 0; t < 4; ++t) o[t][r] *= alpha;
      const int prow = r + ((lane >> 4) << 3);
      Ps[wid][prow][(lane & 15)]      = (__bf16)p0;
      Ps[wid][prow][16 + (lane & 15)] = (__bf16)p1;
    }

    // O += P(16x32) @ V(32x64); LDS ops from same wave stay in order
    v16bf pa = load_frag(&Ps[wid][(lane & 15)][0], lane);
#pragma unroll
    for (int t = 0; t < 4; ++t) {
      v16bf vb = load_frag(&Vt[t * 16 + (lane & 15)][0], lane);
      o[t] = wmma_bf16(pa, vb, o[t]);
    }
  }

  // finalize: divide by l, write [B,N,H*D] bf16 (contiguous rows for kernel 3)
#pragma unroll
  for (int t = 0; t < 4; ++t) {
#pragma unroll
    for (int r = 0; r < 8; ++r) {
      const int qr = qBase + wid * 16 + r + ((lane >> 4) << 3);
      const float val = o[t][r] * (1.0f / lrow[r]);
      aows[((size_t)bb * kN + qr) * kC + hh * kD + t * 16 + (lane & 15)] =
          (__bf16)val;
    }
  }
}

// ---------------------------------------------------------------------------
// Kernel 3: out = attn_out(bf16) @ w_proj^T + b_proj, fp32 output.
// Same 128x128 WMMA tiling as kernel 1; A is already bf16.
// ---------------------------------------------------------------------------
__global__ __launch_bounds__(256)
void proj_kernel(const __bf16* __restrict__ a, const float* __restrict__ w,
                 const float* __restrict__ bias, float* __restrict__ out) {
  __shared__ __align__(16) __bf16 As[128][40];
  __shared__ __align__(16) __bf16 Bs[128][40];

  const int tid  = threadIdx.x;
  const int lane = tid & 31, wid = tid >> 5;
  const int wm = wid & 3, wn = wid >> 2;
  const int rowBase = blockIdx.x * 128;
  const int colBase = blockIdx.y * 128;
  const int K = kC;

  v8f acc[2][4] = {};

  for (int k0 = 0; k0 < K; k0 += 32) {
    {  // stage A (bf16 direct), 16 elems / thread
      const int row = tid >> 1;
      const int c16 = (tid & 1) << 4;
      const __bf16* src = a + (size_t)(rowBase + row) * K + k0 + c16;
      *(v8bf*)&As[row][c16]     = *(const v8bf*)(src);
      *(v8bf*)&As[row][c16 + 8] = *(const v8bf*)(src + 8);
      if (k0 + 32 < K) __builtin_prefetch(src + 32, 0, 1);
    }
    stage_f32(Bs, w + (size_t)colBase * K, K, k0, tid, k0 + 32 < K);
    __syncthreads();
    v16bf af[2], bf[4];
#pragma unroll
    for (int i = 0; i < 2; ++i)
      af[i] = load_frag(&As[wm * 32 + i * 16 + (lane & 15)][0], lane);
#pragma unroll
    for (int j = 0; j < 4; ++j)
      bf[j] = load_frag(&Bs[wn * 64 + j * 16 + (lane & 15)][0], lane);
#pragma unroll
    for (int i = 0; i < 2; ++i)
#pragma unroll
      for (int j = 0; j < 4; ++j)
        acc[i][j] = wmma_bf16(af[i], bf[j], acc[i][j]);
    __syncthreads();
  }

  const int hhalf = (lane >> 4) << 3;
#pragma unroll
  for (int i = 0; i < 2; ++i) {
#pragma unroll
    for (int j = 0; j < 4; ++j) {
      const int cg = colBase + wn * 64 + j * 16 + (lane & 15);
#pragma unroll
      for (int r = 0; r < 8; ++r) {
        const int mg = rowBase + wm * 32 + i * 16 + r + hhalf;
        out[(size_t)mg * kC + cg] = acc[i][j][r] + bias[cg];
      }
    }
  }
}

// ---------------------------------------------------------------------------
extern "C" void kernel_launch(void* const* d_in, const int* in_sizes, int n_in,
                              void* d_out, int out_size, void* d_ws,
                              size_t ws_size, hipStream_t stream) {
  const float* x      = (const float*)d_in[0];
  const float* w_qkv  = (const float*)d_in[1];
  const float* b_qkv  = (const float*)d_in[2];
  const float* w_proj = (const float*)d_in[3];
  const float* b_proj = (const float*)d_in[4];
  const float* fcos   = (const float*)d_in[5];
  const float* fsin   = (const float*)d_in[6];

  const size_t qkvElems = (size_t)kB * kH * kN * kD;   // 4M bf16 each
  __bf16* qws  = (__bf16*)d_ws;
  __bf16* kws  = qws + qkvElems;
  __bf16* vws  = kws + qkvElems;
  __bf16* aows = vws + qkvElems;                       // [B,N,C] bf16
  // total workspace: 4 * 8 MB = 32 MB

  dim3 g1(kM / 128, (3 * kC) / 128);                   // 32 x 24 = 768 blocks
  qkv_rope_kernel<<<g1, 256, 0, stream>>>(x, w_qkv, b_qkv, fcos, fsin,
                                          qws, kws, vws);

  dim3 g2(kN / 128, kB * kH);                          // 16 x 32 = 512 blocks
  flash_attn_kernel<<<g2, 256, 0, stream>>>(qws, kws, vws, aows);

  dim3 g3(kM / 128, kC / 128);                         // 32 x 8 = 256 blocks
  proj_kernel<<<g3, 256, 0, stream>>>(aows, w_proj, b_proj, (float*)d_out);
}
